// MLP_2HLSkipWithEideticMem_29076928593992
// MI455X (gfx1250) — matrix operand order repaired and checked by reading, not compiled, not run on hardware
//
#include <hip/hip_runtime.h>

// ---------------------------------------------------------------------------
// MLP with eidetic (nearest-neighbor) memory, CDNA5 / gfx1250 (wave32, WMMA)
// B=512, D_SENS=784, D_IDX=64, D_INT=32, D_OUT=10, M=262144
//
// Distance GEMM uses split-bf16 (hi/lo) v_wmma_f32_16x16x32_bf16:
//   dot ~= aHi*kHi + aHi*kLo + aLo*kHi  (~fp32 accuracy, bf16 matrix rate)
// Keys are converted fp32->bf16 hi/lo exactly once per key, staged in LDS.
// Three independent WMMA accumulator chains hide XDL latency.
// ---------------------------------------------------------------------------

#define BATCH   512
#define D_SENS  784
#define D_IDX   64
#define D_INT   32
#define D_OUT   10
#define MKEYS   262144

typedef __attribute__((ext_vector_type(16))) __bf16 v16bf;
typedef __attribute__((ext_vector_type(8)))  float  v8f;

__device__ __forceinline__ unsigned short bfbits(__bf16 b) {
    union { __bf16 b; unsigned short s; } u; u.b = b; return u.s;
}

// float -> monotonic unsigned (preserves total order including negatives)
__device__ __forceinline__ unsigned f2mono(float f) {
    unsigned u = __float_as_uint(f);
    return (u & 0x80000000u) ? ~u : (u | 0x80000000u);
}

// ---------------------------------------------------------------------------
// Kernel 0: init argmin slots
// ---------------------------------------------------------------------------
__global__ void init_best_kernel(unsigned long long* best) {
    int i = blockIdx.x * blockDim.x + threadIdx.x;
    if (i < BATCH) best[i] = ~0ull;
}

// ---------------------------------------------------------------------------
// Kernel 1: indexer  a_idx = relu(x @ W1^T + b1)   [512 x 64]
// Also emits the bf16 hi/lo planes of a_idx for the WMMA distance kernel.
// grid = 512 (batch row), block = 64 (output feature)
// ---------------------------------------------------------------------------
__global__ void indexer_kernel(const float* __restrict__ x,
                               const float* __restrict__ W1,
                               const float* __restrict__ b1,
                               float* __restrict__ aidx,
                               unsigned short* __restrict__ ahi,
                               unsigned short* __restrict__ alo) {
    const int j = threadIdx.x;   // 0..63
    const int b = blockIdx.x;    // 0..511
    const float* xr = x  + (size_t)b * D_SENS;
    const float* wr = W1 + (size_t)j * D_SENS;
    float acc = 0.0f;
    #pragma unroll 4
    for (int k = 0; k < D_SENS; ++k) acc = fmaf(xr[k], wr[k], acc);
    acc += b1[j];
    const float a = fmaxf(acc, 0.0f);
    aidx[b * D_IDX + j] = a;
    const __bf16 h = (__bf16)a;
    ahi[b * D_IDX + j] = bfbits(h);
    alo[b * D_IDX + j] = bfbits((__bf16)(a - (float)h));
}

// ---------------------------------------------------------------------------
// Kernel 2: fused key-split + distance GEMM (bf16 WMMA) + argmin merge
//
// grid = 512 WGs x 256 threads (8 waves). Each WG owns 512 consecutive keys.
// Stage 1: convert fp32 keys -> bf16 hi/lo planes in LDS (once per key),
//          computing exact fp32 squared norms on the fly.
// Stage 2: each wave owns 64 keys (4 chunks of 16 columns); loops over the
//          32 batch tiles doing 6 WMMAs per 16x16 tile (3 interleaved
//          accumulator chains), then a shfl_xor argmin and one packed u64
//          atomicMin per row.
// ---------------------------------------------------------------------------
#define DIST_WGS  512
#define WG_SPAN   (MKEYS / DIST_WGS)   // 512 keys per WG
#define KROW      72                    // padded LDS row (72 bf16 = 144 B)

__global__ void __launch_bounds__(256)
distance_argmin_kernel(const float* __restrict__ keys,             // [M][64] fp32
                       const unsigned short* __restrict__ ahi,     // [512][64] bf16 bits
                       const unsigned short* __restrict__ alo,     // [512][64] bf16 bits
                       unsigned long long* __restrict__ best) {    // [512]
    __shared__ __align__(16) unsigned short s_khi[WG_SPAN * KROW]; // 72 KB
    __shared__ __align__(16) unsigned short s_klo[WG_SPAN * KROW]; // 72 KB
    __shared__ float s_nrm[WG_SPAN];                               // 2 KB

    const int t      = threadIdx.x;
    const int wgBase = blockIdx.x * WG_SPAN;

    // ---------------- stage 1: key split into LDS (once per key) ----------
    #pragma unroll 1
    for (int p = 0; p < WG_SPAN / 64; ++p) {
        const int r   = p * 64 + (t >> 2);  // local key row 0..511
        const int seg = t & 3;              // 16-float segment of the row
        const float* src = keys + (size_t)(wgBase + r) * D_IDX + seg * 16;
        float4 f4[4];
        f4[0] = ((const float4*)src)[0];
        f4[1] = ((const float4*)src)[1];
        f4[2] = ((const float4*)src)[2];
        f4[3] = ((const float4*)src)[3];
        const float* v = (const float*)f4;

        unsigned hq[8], lq[8];
        float ss = 0.0f;
        #pragma unroll
        for (int i = 0; i < 8; ++i) {
            const float f0 = v[2 * i], f1 = v[2 * i + 1];
            ss += f0 * f0 + f1 * f1;
            const __bf16 h0 = (__bf16)f0, h1 = (__bf16)f1;
            const __bf16 l0 = (__bf16)(f0 - (float)h0);
            const __bf16 l1 = (__bf16)(f1 - (float)h1);
            hq[i] = (unsigned)bfbits(h0) | ((unsigned)bfbits(h1) << 16);
            lq[i] = (unsigned)bfbits(l0) | ((unsigned)bfbits(l1) << 16);
        }
        unsigned short* dh = s_khi + r * KROW + seg * 16;
        ((uint4*)dh)[0] = make_uint4(hq[0], hq[1], hq[2], hq[3]);
        ((uint4*)dh)[1] = make_uint4(hq[4], hq[5], hq[6], hq[7]);
        unsigned short* dl = s_klo + r * KROW + seg * 16;
        ((uint4*)dl)[0] = make_uint4(lq[0], lq[1], lq[2], lq[3]);
        ((uint4*)dl)[1] = make_uint4(lq[4], lq[5], lq[6], lq[7]);

        // exact fp32 norm: reduce the 4 segment threads (adjacent lanes)
        ss += __shfl_xor(ss, 1, 32);
        ss += __shfl_xor(ss, 2, 32);
        if (seg == 0) s_nrm[r] = ss;
    }
    __syncthreads();

    // ---------------- stage 2: WMMA distance + argmin ----------------------
    const int lane = t & 31;
    const int wave = t >> 5;
    const int half = lane >> 4;   // 0: lanes 0-15, 1: lanes 16-31
    const int l16  = lane & 15;

    union V16 { v16bf v; uint4 q[2]; };

    // this wave's first local key row and its fixed LDS column offset
    const int waveRow0 = wave * 64 + l16;
    const int colOff   = half * 16;                 // elements

    #pragma clang loop unroll(disable)
    for (int bt = 0; bt < BATCH / 16; ++bt) {
        // A operand, contiguous k order (same as B):
        //   lanes 0-15 : M=lane,    K = s*32 + 0..15
        //   lanes 16-31: M=lane-16, K = s*32 + 16..31
        const int arow = bt * 16 + l16;
        v16bf aH[2], aL[2];
        #pragma unroll
        for (int s = 0; s < 2; ++s) {
            const size_t eb = (size_t)arow * D_IDX + s * 32 + half * 16;
            V16 u;
            u.q[0] = *(const uint4*)(ahi + eb);
            u.q[1] = *(const uint4*)(ahi + eb + 8);
            aH[s] = u.v;
            u.q[0] = *(const uint4*)(alo + eb);
            u.q[1] = *(const uint4*)(alo + eb + 8);
            aL[s] = u.v;
        }

        float    bd[8];
        unsigned bi[8];
        #pragma unroll
        for (int r = 0; r < 8; ++r) { bd[r] = 3.4e38f; bi[r] = 0u; }

        int rowBase = waveRow0;                     // local key row, +16/chunk
        #pragma clang loop unroll(disable)
        for (int ch = 0; ch < 4; ++ch) {
            // B operand, contiguous k order (same as A):
            //   lanes 0-15 : N=lane,    K = s*32 + 0..15
            //   lanes 16-31: N=lane-16, K = s*32 + 16..31
            const int eo = rowBase * KROW + colOff; // running element offset
            v16bf bH[2], bL[2];
            #pragma unroll
            for (int s = 0; s < 2; ++s) {
                V16 u;
                u.q[0] = *(const uint4*)(s_khi + eo + s * 32);
                u.q[1] = *(const uint4*)(s_khi + eo + s * 32 + 8);
                bH[s] = u.v;
                u.q[0] = *(const uint4*)(s_klo + eo + s * 32);
                u.q[1] = *(const uint4*)(s_klo + eo + s * 32 + 8);
                bL[s] = u.v;
            }

            // three independent accumulator chains, interleaved so each
            // dependent WMMA is 3 instructions behind its producer
            v8f c0 = {}, c1 = {}, c2 = {};
            c0 = __builtin_amdgcn_wmma_f32_16x16x32_bf16(false, aH[0], false, bH[0], (short)0, c0, false, false);
            c1 = __builtin_amdgcn_wmma_f32_16x16x32_bf16(false, aH[0], false, bL[0], (short)0, c1, false, false);
            c2 = __builtin_amdgcn_wmma_f32_16x16x32_bf16(false, aL[0], false, bH[0], (short)0, c2, false, false);
            c0 = __builtin_amdgcn_wmma_f32_16x16x32_bf16(false, aH[1], false, bH[1], (short)0, c0, false, false);
            c1 = __builtin_amdgcn_wmma_f32_16x16x32_bf16(false, aH[1], false, bL[1], (short)0, c1, false, false);
            c2 = __builtin_amdgcn_wmma_f32_16x16x32_bf16(false, aL[1], false, bH[1], (short)0, c2, false, false);

            const float    nrm = s_nrm[rowBase];
            const unsigned key = (unsigned)(wgBase + rowBase);
            #pragma unroll
            for (int r = 0; r < 8; ++r) {
                const float dot = c0[r] + c1[r] + c2[r];
                const float d   = nrm - 2.0f * dot; // row bt*16+r+half*8
                if (d < bd[r]) { bd[r] = d; bi[r] = key; }
            }
            rowBase += 16;
        }

        // reduce across the 16 columns (stays within each 16-lane half)
        #pragma unroll
        for (int r = 0; r < 8; ++r) {
            float    d = bd[r];
            unsigned i = bi[r];
            #pragma unroll
            for (int m = 8; m >= 1; m >>= 1) {
                const float    od = __shfl_xor(d, m, 32);
                const unsigned oi = __shfl_xor(i, m, 32);
                if (od < d) { d = od; i = oi; }
            }
            if (l16 == 0) {
                const int row = bt * 16 + r + half * 8;
                const unsigned long long pk =
                    ((unsigned long long)f2mono(d) << 32) | (unsigned long long)i;
                atomicMin(best + row, pk);
            }
        }
    }
}

// ---------------------------------------------------------------------------
// Kernel 3: gather + integrator + head.  One wave per batch row.
// a_int = relu(a_idx@W2^T + b2 + mem_values[nn]@W3^T + b3); out = a_int@W4^T+b4
// ---------------------------------------------------------------------------
__global__ void head_kernel(const float* __restrict__ aidx,
                            const unsigned long long* __restrict__ best,
                            const float* __restrict__ memv,
                            const float* __restrict__ W2, const float* __restrict__ b2,
                            const float* __restrict__ W3, const float* __restrict__ b3,
                            const float* __restrict__ W4, const float* __restrict__ b4,
                            float* __restrict__ out) {
    const int row = blockIdx.x;   // 0..511
    const int j   = threadIdx.x;  // 0..31

    const unsigned nn = (unsigned)(best[row] & 0xFFFFFFFFull);
    const float* xr = memv + (size_t)nn * D_SENS;
    const float* ar = aidx + (size_t)row * D_IDX;
    const float* w2 = W2 + (size_t)j * D_IDX;
    const float* w3 = W3 + (size_t)j * D_SENS;

    float acc = b2[j] + b3[j];
    #pragma unroll 4
    for (int k = 0; k < D_IDX;  ++k) acc = fmaf(ar[k], w2[k], acc);
    #pragma unroll 4
    for (int k = 0; k < D_SENS; ++k) acc = fmaf(xr[k], w3[k], acc);
    const float aint = fmaxf(acc, 0.0f);

    __shared__ float s[D_INT];
    s[j] = aint;
    __syncthreads();

    if (j < D_OUT) {
        const float* w4 = W4 + (size_t)j * D_INT;
        float o = b4[j];
        #pragma unroll
        for (int k = 0; k < D_INT; ++k) o = fmaf(s[k], w4[k], o);
        out[row * D_OUT + j] = o;
    }
}

// ---------------------------------------------------------------------------
// Launcher
// ---------------------------------------------------------------------------
extern "C" void kernel_launch(void* const* d_in, const int* in_sizes, int n_in,
                              void* d_out, int out_size, void* d_ws, size_t ws_size,
                              hipStream_t stream) {
    const float* x    = (const float*)d_in[0];
    const float* W1   = (const float*)d_in[1];
    const float* b1   = (const float*)d_in[2];
    const float* W2   = (const float*)d_in[3];
    const float* b2   = (const float*)d_in[4];
    const float* W3   = (const float*)d_in[5];
    const float* b3   = (const float*)d_in[6];
    const float* W4   = (const float*)d_in[7];
    const float* b4   = (const float*)d_in[8];
    const float* keys = (const float*)d_in[9];
    const float* memv = (const float*)d_in[10];
    float* out = (float*)d_out;

    // workspace layout (~266 KB total)
    char* ws = (char*)d_ws;
    float*              aidx = (float*)ws;                                  // 512*64 f32
    unsigned short*     ahi  = (unsigned short*)(ws + 131072);              // 512*64 bf16
    unsigned short*     alo  = (unsigned short*)(ws + 131072 + 65536);      // 512*64 bf16
    unsigned long long* best = (unsigned long long*)(ws + 131072 + 131072); // 512 u64

    init_best_kernel<<<(BATCH + 255) / 256, 256, 0, stream>>>(best);
    indexer_kernel<<<BATCH, D_IDX, 0, stream>>>(x, W1, b1, aidx, ahi, alo);
    distance_argmin_kernel<<<DIST_WGS, 256, 0, stream>>>(keys, ahi, alo, best);
    head_kernel<<<BATCH, D_INT, 0, stream>>>(aidx, best, memv, W2, b2, W3, b3, W4, b4, out);
}
